// VAT_self_attention_1D_89343909691495
// MI455X (gfx1250) — matrix operand order repaired and checked by reading, not compiled, not run on hardware
//
#include <hip/hip_runtime.h>

// ---------------------------------------------------------------------------
// Sizes (fixed by the reference)
// ---------------------------------------------------------------------------
static constexpr int B   = 4;
static constexpr int T   = 4096;
static constexpr int FIN = 229;
static constexpr int FP  = 256;     // Fin padded to 256 (multiple of 32)
static constexpr int D   = 256;
static constexpr int G   = 8;
static constexpr int DG  = 32;
static constexpr int KW  = 31;      // window
static constexpr int PAD = 15;      // (KW-1)/2
static constexpr int O   = 88;
static constexpr int OP  = 96;      // O padded to 96 (6 tiles of 16)
static constexpr int TP  = T + 64;  // padded time rows for k/v slabs
static constexpr float LN_EPS = 1e-5f;

// workspace layout (bytes)
static constexpr size_t OFF_SPECB = 0;                                   // [B][T][FP] bf16
static constexpr size_t SZ_SPECB  = (size_t)B * T * FP * 2;
static constexpr size_t OFF_WQKV  = OFF_SPECB + SZ_SPECB;                // [3][D][FP] bf16
static constexpr size_t SZ_WQKV   = (size_t)3 * D * FP * 2;
static constexpr size_t OFF_LINW  = OFF_WQKV + SZ_WQKV;                  // [OP][D] bf16
static constexpr size_t SZ_LINW   = (size_t)OP * D * 2;
static constexpr size_t OFF_QB    = OFF_LINW + SZ_LINW;                  // [B][T][D] bf16
static constexpr size_t SZ_QB     = (size_t)B * T * D * 2;
static constexpr size_t OFF_KPAD  = OFF_QB + SZ_QB;                      // [B][TP][D] bf16
static constexpr size_t SZ_KPAD   = (size_t)B * TP * D * 2;
static constexpr size_t OFF_VPAD  = OFF_KPAD + SZ_KPAD;                  // [B][TP][D] bf16
static constexpr size_t SZ_VPAD   = SZ_KPAD;
static constexpr size_t OFF_RELB  = OFF_VPAD + SZ_VPAD;                  // [G][2][32][16] bf16
static constexpr size_t SZ_RELB   = (size_t)G * 2 * 32 * 16 * 2;

// ---------------------------------------------------------------------------
// Vector types / WMMA helper
// ---------------------------------------------------------------------------
typedef __attribute__((ext_vector_type(16))) __bf16          v16bf;
typedef __attribute__((ext_vector_type(16))) unsigned short  v16u;
typedef __attribute__((ext_vector_type(8)))  float           v8f;

__device__ __forceinline__ unsigned short f2bf(float f) {
    union { float f; unsigned u; } x; x.f = f;
    unsigned r = x.u + 0x7FFFu + ((x.u >> 16) & 1u);   // round-nearest-even
    return (unsigned short)(r >> 16);
}

__device__ __forceinline__ v8f wmma_bf16(v16bf a, v16bf b, v8f c) {
    return __builtin_amdgcn_wmma_f32_16x16x32_bf16(false, a, false, b,
                                                   (short)0, c, false, false);
}

// A-fragment (16x32 bf16, row-major source): lane<16 row=lane, K = {0..15};
// lane>=16 row=lane-16, K = {16..31}
__device__ __forceinline__ v16bf load_afrag(const unsigned short* base, int stride, int lane) {
    const unsigned short* p = base + (size_t)(lane & 15) * stride + ((lane >> 4) << 4);
    v16u u;
#pragma unroll
    for (int e = 0; e < 16; ++e) u[e] = p[e];
    return __builtin_bit_cast(v16bf, u);
}

// B-fragment from an n-major source (row = column index n, contiguous = K):
// lane n = lane&15, K = (lane>>4)*16 + e contiguous in memory.
__device__ __forceinline__ v16bf load_bfrag(const unsigned short* base, int stride, int lane) {
    const unsigned short* p = base + (size_t)(lane & 15) * stride + ((lane >> 4) << 4);
    v16u u;
#pragma unroll
    for (int e = 0; e < 16; ++e) u[e] = p[e];
    return __builtin_bit_cast(v16bf, u);
}

// contiguous 16-element per-lane fragment (pre-swizzled tables, e.g. relb)
__device__ __forceinline__ v16bf load_packed(const unsigned short* p) {
    v16u u;
#pragma unroll
    for (int e = 0; e < 16; ++e) u[e] = p[e];
    return __builtin_bit_cast(v16bf, u);
}

// ---------------------------------------------------------------------------
// Prep kernels: fp32 -> bf16 with zero padding
// ---------------------------------------------------------------------------
__global__ void cvt_spec(const float* __restrict__ spec, unsigned short* __restrict__ dst) {
    size_t i = (size_t)blockIdx.x * blockDim.x + threadIdx.x;   // over B*T*FP
    if (i >= (size_t)B * T * FP) return;
    int f = (int)(i & (FP - 1));
    size_t bt = i >> 8;
    dst[i] = (f < FIN) ? f2bf(spec[bt * FIN + f]) : (unsigned short)0;
}

__global__ void cvt_w(const float* __restrict__ wq, const float* __restrict__ wk,
                      const float* __restrict__ wv, unsigned short* __restrict__ dst) {
    size_t i = (size_t)blockIdx.x * blockDim.x + threadIdx.x;   // over 3*D*FP
    if (i >= (size_t)3 * D * FP) return;
    int f = (int)(i & (FP - 1));
    size_t nd = i >> 8;
    int n = (int)(nd % D);
    int m = (int)(nd / D);
    const float* src = (m == 0) ? wq : (m == 1) ? wk : wv;
    dst[i] = (f < FIN) ? f2bf(src[(size_t)n * FIN + f]) : (unsigned short)0;
}

__global__ void cvt_linw(const float* __restrict__ lw, unsigned short* __restrict__ dst) {
    size_t i = (size_t)blockIdx.x * blockDim.x + threadIdx.x;   // over OP*D
    if (i >= (size_t)OP * D) return;
    int d = (int)(i & (D - 1));
    int o = (int)(i >> 8);
    dst[i] = (o < O) ? f2bf(lw[(size_t)o * D + d]) : (unsigned short)0;
}

// rel (D,KW) fp32 -> per-lane pre-swizzled bf16 B-fragments:
// relb[((g*2 + kt)*32 + lane)*16 + e] with n=k=kt*16+(lane&15), K=d=(lane>>4)*16+e
__global__ void cvt_rel(const float* __restrict__ rel, unsigned short* __restrict__ dst) {
    int i = blockIdx.x * blockDim.x + threadIdx.x;              // over G*2*32*16 = 8192
    if (i >= G * 2 * 32 * 16) return;
    int e    = i & 15;
    int lane = (i >> 4) & 31;
    int kt   = (i >> 9) & 1;
    int g    = i >> 10;
    int k  = kt * 16 + (lane & 15);
    int dg = ((lane >> 4) << 4) + e;
    dst[i] = (k < KW) ? f2bf(rel[(size_t)(g * DG + dg) * KW + k]) : (unsigned short)0;
}

// ---------------------------------------------------------------------------
// Projections: q = spec@Wq^T ; k/v into zero-padded slabs (row offset PAD)
// one wave per (matrix, 16-row tile); 8 K-steps x 16 N-tiles of WMMA
// ---------------------------------------------------------------------------
__global__ void proj_kernel(const unsigned short* __restrict__ specb,
                            const unsigned short* __restrict__ wqkv,
                            unsigned short* __restrict__ qb,
                            unsigned short* __restrict__ kpad,
                            unsigned short* __restrict__ vpad) {
    int lane = threadIdx.x & 31;
    int wid  = blockIdx.x * (blockDim.x >> 5) + (threadIdx.x >> 5);
    const int tilesPerMat = (B * T) / 16;           // 1024
    if (wid >= 3 * tilesPerMat) return;
    int m  = wid / tilesPerMat;                     // 0=q 1=k 2=v
    int rt = wid - m * tilesPerMat;
    int b  = rt >> 8;                               // 256 tiles per batch
    int t0 = (rt & 255) << 4;

    const unsigned short* A0 = specb + ((size_t)b * T + t0) * FP;
    v16bf a[8];
#pragma unroll
    for (int kk = 0; kk < 8; ++kk) a[kk] = load_afrag(A0 + kk * 32, FP, lane);

    // wave-uniform destination selection, hoisted out of the store loops
    unsigned short* dstBase;
    if (m == 0)      dstBase = qb   + ((size_t)b * T  + t0) * D;
    else if (m == 1) dstBase = kpad + ((size_t)b * TP + t0 + PAD) * D;
    else             dstBase = vpad + ((size_t)b * TP + t0 + PAD) * D;

    const unsigned short* Wm = wqkv + (size_t)m * D * FP;
    int Mbase = (lane >> 4) << 3;
    for (int nt = 0; nt < 16; ++nt) {
        const unsigned short* Wt = Wm + (size_t)(nt * 16) * FP;
        v8f acc = {};
#pragma unroll
        for (int kk = 0; kk < 8; ++kk) {
            v16bf bf = load_bfrag(Wt + kk * 32, FP, lane);
            acc = wmma_bf16(a[kk], bf, acc);
        }
        int n = nt * 16 + (lane & 15);
#pragma unroll
        for (int r = 0; r < 8; ++r)
            dstBase[(size_t)(Mbase + r) * D + n] = f2bf(acc[r]);
    }
}

// ---------------------------------------------------------------------------
// Fused attention + softmax + LayerNorm + linear + sigmoid
// 1 block = 16 time steps; wave g handles group g
// ---------------------------------------------------------------------------
__global__ void attn_kernel(const unsigned short* __restrict__ qb,
                            const unsigned short* __restrict__ kpad,
                            const unsigned short* __restrict__ vpad,
                            const unsigned short* __restrict__ relb,
                            const float* __restrict__ ln_gamma,
                            const float* __restrict__ ln_beta,
                            const unsigned short* __restrict__ linwb,
                            const float* __restrict__ lin_b,
                            float* __restrict__ frame_pred,
                            float* __restrict__ attn_out) {
    __shared__ __align__(16) unsigned char lds[57344];
    // per-wave slot g*5120 .. : energies [16][80] f32 (cols 0..47 band E', 48..79 rel E)
    //   -> after S3 the same slot is reused as Vt [32][64] bf16 (transposed V slab)
    // 40960..57344: per-wave banded A' [16][64] bf16
    // after S4a: sOut [16][264] f32 at 0, sH [16][256] bf16 at 20480
    int lane = threadIdx.x & 31;
    int g    = threadIdx.x >> 5;
    int b    = blockIdx.x >> 8;
    int t0   = (blockIdx.x & 255) << 4;

    float*          Ew   = (float*)(lds + (size_t)g * 5120);        // [16][80]
    unsigned short* Vt   = (unsigned short*)(lds + (size_t)g * 5120); // [32][64] (after S3)
    unsigned short* Apg  = (unsigned short*)(lds + 40960) + (size_t)g * 16 * 64;
    float*          sOut = (float*)lds;                             // [16][264]
    unsigned short* sH   = (unsigned short*)(lds + 20480);          // [16][256]

    int Mbase = (lane >> 4) << 3;
    int nlo   = lane & 15;

    const unsigned short* vsrc = vpad + ((size_t)b * TP + t0) * D + g * DG;
    // early prefetch of the V slab we will stage after softmax
    __builtin_prefetch(vsrc, 0, 1);
    __builtin_prefetch(vsrc + (size_t)32 * D, 0, 1);

    // Q fragment: 16 rows x 32 (Dg) bf16 -> exactly one A frag
    v16bf qa = load_afrag(qb + ((size_t)b * T + t0) * D + g * DG, D, lane);

    // band energies E'[t][j] = q[t] . kpad[t0+j],  j = 0..47
#pragma unroll
    for (int jt = 0; jt < 3; ++jt) {
        v16bf bf = load_bfrag(kpad + ((size_t)b * TP + t0 + jt * 16) * D + g * DG, D, lane);
        v8f acc = {};
        acc = wmma_bf16(qa, bf, acc);
        int col = jt * 16 + nlo;
#pragma unroll
        for (int r = 0; r < 8; ++r) Ew[(Mbase + r) * 80 + col] = acc[r];
    }
    // rel energies Er[t][k] = q[t] . rel_g[:,k]  (pre-swizzled bf16 fragments)
#pragma unroll
    for (int kt = 0; kt < 2; ++kt) {
        v16bf bf = load_packed(relb + (size_t)((g * 2 + kt) * 32 + lane) * 16);
        v8f acc = {};
        acc = wmma_bf16(qa, bf, acc);
        int col = 48 + kt * 16 + nlo;
#pragma unroll
        for (int r = 0; r < 8; ++r) Ew[(Mbase + r) * 80 + col] = acc[r];
    }
    __syncthreads();   // S1: energies visible

    // zero banded A'
    for (int j = lane; j < 16 * 64; j += 32) Apg[j] = 0;
    __syncthreads();   // S2

    // softmax per row (lane < 16 owns row t=lane), write attn probs + banded A'
    if (lane < 16) {
        int t = lane;
        float e[KW];
        float mx = -3.0e38f;
#pragma unroll
        for (int k = 0; k < KW; ++k) {
            e[k] = Ew[t * 80 + t + k] + Ew[t * 80 + 48 + k];
            mx = fmaxf(mx, e[k]);
        }
        float s = 0.f;
#pragma unroll
        for (int k = 0; k < KW; ++k) { e[k] = __expf(e[k] - mx); s += e[k]; }
        float inv = 1.f / s;
        size_t abase = (((size_t)b * T + t0 + t) * G + g) * KW;
#pragma unroll
        for (int k = 0; k < KW; ++k) {
            float p = e[k] * inv;
            attn_out[abase + k] = p;
            Apg[t * 64 + t + k] = f2bf(p);
        }
    }
    __syncthreads();   // S3: A' ready; energy region dead -> reuse as Vt

    // banded A' fragments (per-wave LDS region)
    v16bf ap0 = load_afrag(Apg + 0,  64, lane);
    v16bf ap1 = load_afrag(Apg + 32, 64, lane);

    // stage V slab transposed into own slot: Vt[d][j] = vpad[t0+j][g*32+d]
    // coalesced: lanes sweep contiguous d, 2 rows per iteration
    for (int jj = 0; jj < 64; jj += 2) {
        int j  = jj + (lane >> 4);
        int d2 = (lane & 15) * 2;
        const unsigned short* p = vsrc + (size_t)j * D + d2;
        unsigned short a0 = p[0];
        unsigned short a1 = p[1];
        Vt[(d2)     * 64 + j] = a0;
        Vt[(d2 + 1) * 64 + j] = a1;
    }
    __syncthreads();   // S3b: Vt visible

    // out = A'(16x64) @ Vslab(64x32): 2 K-steps x 2 N-tiles, B frags from LDS Vt
    v8f oacc[2];
#pragma unroll
    for (int nt = 0; nt < 2; ++nt) {
        v8f acc = {};
        v16bf b0 = load_bfrag(Vt + (size_t)(nt * 16) * 64 + 0,  64, lane);
        acc = wmma_bf16(ap0, b0, acc);
        v16bf b1 = load_bfrag(Vt + (size_t)(nt * 16) * 64 + 32, 64, lane);
        acc = wmma_bf16(ap1, b1, acc);
        oacc[nt] = acc;
    }
    __syncthreads();   // S4a: Vt dead; sOut region safe to write

#pragma unroll
    for (int nt = 0; nt < 2; ++nt) {
        int col = g * DG + nt * 16 + nlo;
#pragma unroll
        for (int r = 0; r < 8; ++r) sOut[(Mbase + r) * 264 + col] = oacc[nt][r];
    }
    __syncthreads();   // S4: full 16x256 out tile in LDS

    // LayerNorm: wave g handles rows 2g, 2g+1
#pragma unroll
    for (int rr = 0; rr < 2; ++rr) {
        int r = g * 2 + rr;
        float x[8];
        float s = 0.f;
#pragma unroll
        for (int i = 0; i < 8; ++i) { x[i] = sOut[r * 264 + lane + 32 * i]; s += x[i]; }
#pragma unroll
        for (int off = 16; off >= 1; off >>= 1) s += __shfl_xor(s, off);
        float mu = s * (1.f / 256.f);
        float v = 0.f;
#pragma unroll
        for (int i = 0; i < 8; ++i) { float d = x[i] - mu; v += d * d; }
#pragma unroll
        for (int off = 16; off >= 1; off >>= 1) v += __shfl_xor(v, off);
        float inv = rsqrtf(v * (1.f / 256.f) + LN_EPS);
#pragma unroll
        for (int i = 0; i < 8; ++i) {
            int c = lane + 32 * i;
            float h = (x[i] - mu) * inv * ln_gamma[c] + ln_beta[c];
            sH[r * 256 + c] = f2bf(h);
        }
    }
    __syncthreads();   // S5: h tile ready

    // final linear + sigmoid: waves 0..5 each own one 16-col tile of OP=96
    if (g < 6) {
        v8f acc = {};
#pragma unroll
        for (int kk = 0; kk < 8; ++kk) {
            v16bf ha = load_afrag(sH + kk * 32, 256, lane);
            v16bf bw = load_bfrag(linwb + (size_t)(g * 16) * D + kk * 32, D, lane);
            acc = wmma_bf16(ha, bw, acc);
        }
        int n = g * 16 + nlo;
        if (n < O) {
            float bias = lin_b[n];
#pragma unroll
            for (int r = 0; r < 8; ++r) {
                int Mrow = Mbase + r;
                float val = acc[r] + bias;
                float sig = 1.f / (1.f + __expf(-val));
                frame_pred[((size_t)b * T + t0 + Mrow) * O + n] = sig;
            }
        }
    }
}

// ---------------------------------------------------------------------------
// Host launcher
// ---------------------------------------------------------------------------
extern "C" void kernel_launch(void* const* d_in, const int* in_sizes, int n_in,
                              void* d_out, int out_size, void* d_ws, size_t ws_size,
                              hipStream_t stream) {
    const float* spec     = (const float*)d_in[0];
    const float* W_q      = (const float*)d_in[1];
    const float* W_k      = (const float*)d_in[2];
    const float* W_v      = (const float*)d_in[3];
    const float* rel      = (const float*)d_in[4];
    const float* ln_gamma = (const float*)d_in[5];
    const float* ln_beta  = (const float*)d_in[6];
    const float* lin_w    = (const float*)d_in[7];
    const float* lin_b    = (const float*)d_in[8];

    unsigned char* ws = (unsigned char*)d_ws;
    unsigned short* specb = (unsigned short*)(ws + OFF_SPECB);
    unsigned short* wqkv  = (unsigned short*)(ws + OFF_WQKV);
    unsigned short* linwb = (unsigned short*)(ws + OFF_LINW);
    unsigned short* qb    = (unsigned short*)(ws + OFF_QB);
    unsigned short* kp    = (unsigned short*)(ws + OFF_KPAD);
    unsigned short* vp    = (unsigned short*)(ws + OFF_VPAD);
    unsigned short* relb  = (unsigned short*)(ws + OFF_RELB);

    float* frame_pred = (float*)d_out;
    float* attn_out   = frame_pred + (size_t)B * T * O;

    // zero k/v slabs (covers the time padding rows that make edge windows exact)
    hipMemsetAsync(ws + OFF_KPAD, 0, SZ_KPAD + SZ_VPAD, stream);

    {
        size_t n = (size_t)B * T * FP;
        cvt_spec<<<dim3((unsigned)((n + 255) / 256)), dim3(256), 0, stream>>>(spec, specb);
    }
    {
        size_t n = (size_t)3 * D * FP;
        cvt_w<<<dim3((unsigned)((n + 255) / 256)), dim3(256), 0, stream>>>(W_q, W_k, W_v, wqkv);
    }
    {
        size_t n = (size_t)OP * D;
        cvt_linw<<<dim3((unsigned)((n + 255) / 256)), dim3(256), 0, stream>>>(lin_w, linwb);
    }
    {
        size_t n = (size_t)G * 2 * 32 * 16;
        cvt_rel<<<dim3((unsigned)((n + 255) / 256)), dim3(256), 0, stream>>>(rel, relb);
    }
    // 3 matrices x 1024 row-tiles = 3072 waves, 8 waves per 256-thread block
    proj_kernel<<<dim3(384), dim3(256), 0, stream>>>(specb, wqkv, qb, kp, vp);
    // one block per 16-t tile
    attn_kernel<<<dim3(B * (T / 16)), dim3(256), 0, stream>>>(
        qb, kp, vp, relb, ln_gamma, ln_beta, linwb, lin_b, frame_pred, attn_out);
}
